// RNNEncoder_23330262352484
// MI455X (gfx1250) — matrix operand order repaired and checked by reading, not compiled, not run on hardware
//
#include <hip/hip_runtime.h>
#include <hip/hip_bf16.h>

// Problem dims (reference): B=64, T=512, V=32000, E=256, H=512
#define B_  64
#define T_  512
#define E_  256
#define H_  512
#define G4  2048   // 4*H
#define NBLK 128   // persistent recurrence workgroups (4 m-tiles x 32 n-tiles)

typedef __attribute__((ext_vector_type(16))) _Float16     v16h;
typedef __attribute__((ext_vector_type(8)))  float        v8f;
typedef __attribute__((ext_vector_type(8)))  unsigned int v8u;

static __device__ __forceinline__ unsigned int pack2h(float a, float b) {
    _Float16 ha = (_Float16)a, hb = (_Float16)b;
    unsigned short ua = __builtin_bit_cast(unsigned short, ha);
    unsigned short ub = __builtin_bit_cast(unsigned short, hb);
    return (unsigned int)ua | ((unsigned int)ub << 16);
}

static __device__ __forceinline__ v16h as_v16h(v8u r) {
    return __builtin_bit_cast(v16h, r);
}

static __device__ __forceinline__ float sigmoidf_(float x) {
    return 1.0f / (1.0f + __expf(-x));
}

// ---------------------------------------------------------------------------
// Pack fp32 weight matrix W[K, 2048] into the wave32 WMMA-B (16x16x32 f16)
// lane layout: packed index (((c*128 + tile)*32 + lane)*8 + j) holds the f16
// pair (k, k+1) for n = tile*16 + (lane&15), k = c*32 + ((lane>>4)<<4) + 2*j.
// A wave then fetches a 32x16 B-tile as two b128 loads per lane.
// ---------------------------------------------------------------------------
__global__ void pack_w_kernel(const float* __restrict__ W,
                              unsigned int* __restrict__ P, int K) {
    int gid = blockIdx.x * blockDim.x + threadIdx.x;
    int total = (K / 2) * G4;
    if (gid >= total) return;
    int j  = gid & 7;
    int l  = (gid >> 3) & 31;
    int tl = (gid >> 8) & 127;
    int c  = gid >> 15;
    int n  = tl * 16 + (l & 15);
    int k  = c * 32 + ((l >> 4) << 4) + 2 * j;
    P[gid] = pack2h(W[(long long)k * G4 + n], W[(long long)(k + 1) * G4 + n]);
}

__global__ void init_state_kernel(_Float16* __restrict__ hbuf,
                                  int* __restrict__ bar) {
    int i = blockIdx.x * blockDim.x + threadIdx.x;
    if (i < B_ * H_) hbuf[i] = (_Float16)0.0f;
    if (i < T_)      bar[i]  = 0;
}

__global__ void mask_kernel(const int* __restrict__ seq,
                            float* __restrict__ out) {
    int i = blockIdx.x * blockDim.x + threadIdx.x;
    if (i < B_ * T_) out[i] = (seq[i] != 0) ? 1.0f : 0.0f;
}

// ---------------------------------------------------------------------------
// Phase 1: xz[row, :] = f16( emb[seq[row]] @ Wx + b ), rows = B*T = 32768.
// One wave per block computes a 16x64 tile (4 WMMA accumulators share one A).
// A operand gathered per-lane from the embedding table (fp32 -> f16) per the
// contiguous k order this code uses for A and B (not AMD's table order):
//   lane<16 : M=lane,    K = {0..15} + kc
//   lane>=16: M=lane-16, K = {16..31} + kc
// ---------------------------------------------------------------------------
__global__ void __launch_bounds__(32)
xz_kernel(const int* __restrict__ seq, const float* __restrict__ emb,
          const unsigned int* __restrict__ Wxp, const float* __restrict__ bias,
          _Float16* __restrict__ xz) {
    const int lane = threadIdx.x;
    const int m0   = blockIdx.x * 16;   // row tile of B*T
    const int n0g  = blockIdx.y * 64;   // 64-wide column group of 4H
    const int m    = lane & 15;
    const int hi   = lane >> 4;
    const int koff = hi * 16;
    const int row  = m0 + m;
    const long long tok = seq[row];
    const float* __restrict__ X = emb + tok * E_;

    v8f acc[4] = {};
    for (int kc = 0; kc < E_; kc += 32) {
        const float4* p0 = (const float4*)(X + kc + koff);
        float4 q0 = p0[0], q1 = p0[1];
        const float4* p1 = (const float4*)(X + kc + 8 + koff);
        float4 q2 = p1[0], q3 = p1[1];
        v8u ar;
        ar[0] = pack2h(q0.x, q0.y); ar[1] = pack2h(q0.z, q0.w);
        ar[2] = pack2h(q1.x, q1.y); ar[3] = pack2h(q1.z, q1.w);
        ar[4] = pack2h(q2.x, q2.y); ar[5] = pack2h(q2.z, q2.w);
        ar[6] = pack2h(q3.x, q3.y); ar[7] = pack2h(q3.z, q3.w);
        v16h A = as_v16h(ar);
        int c = kc >> 5;
#pragma unroll
        for (int s = 0; s < 4; ++s) {
            int tl = (n0g >> 4) + s;
            const uint4* bp =
                (const uint4*)(Wxp + ((((long long)c * 128 + tl) * 32 + lane) * 8));
            uint4 b0 = bp[0], b1 = bp[1];
            v8u br;
            br[0] = b0.x; br[1] = b0.y; br[2] = b0.z; br[3] = b0.w;
            br[4] = b1.x; br[5] = b1.y; br[6] = b1.z; br[7] = b1.w;
            acc[s] = __builtin_amdgcn_wmma_f32_16x16x32_f16(
                false, A, false, as_v16h(br), (short)0, acc[s], false, false);
        }
    }
    // D/C layout: lane holds col N=(lane&15), rows M=j+8*hi
#pragma unroll
    for (int s = 0; s < 4; ++s) {
        int n = n0g + s * 16 + m;
        float bv = bias[n];
#pragma unroll
        for (int j = 0; j < 8; ++j) {
            int rr = m0 + j + hi * 8;
            xz[(long long)rr * G4 + n] = (_Float16)(acc[s][j] + bv);
        }
    }
}

// ---------------------------------------------------------------------------
// Phase 2: ONE persistent kernel for all 512 steps. 128 waves (one per block)
// each own a 16(batch) x 16(hidden) tile of all four gates.
//   * Wh slice (64 tiles, 64 KB) loaded ONCE into LDS -> per-step B reads are
//     two ds_load_b128 per tile (320 KB/WGP LDS makes this free).
//   * c state lives in registers for the whole sequence (8 cells per lane).
//   * h double-buffered in f16 in workspace; per-step device sync via a
//     per-step arrival counter (release add + acquire spin, agent scope).
// ---------------------------------------------------------------------------
__global__ void __launch_bounds__(32)
lstm_persistent_kernel(const _Float16* __restrict__ xz,
                       const unsigned int* __restrict__ Whp,
                       _Float16* __restrict__ hA, _Float16* __restrict__ hB,
                       float* __restrict__ seq_out, float* __restrict__ hlast,
                       int* __restrict__ bar) {
    __shared__ __align__(16) unsigned int Bs[16 * 4 * 32 * 8];   // 64 KB

    const int lane = threadIdx.x;
    const int blk  = blockIdx.x;        // 0..127
    const int m0   = (blk & 3) * 16;    // batch-row tile
    const int nt   = blk >> 2;          // hidden-column tile index (0..31)
    const int n0   = nt * 16;
    const int m    = lane & 15;
    const int hi   = lane >> 4;
    const int koff = hi * 16;
    const int row  = m0 + m;
    const int hcol = n0 + m;

    // ---- Stage this block's Wh slice into LDS (once) ----
    // LDS index ((c*4+g)*32+l)*8+j  <-  Whp[(((c*128)+(g*32+nt))*32+l)*8+j]
    for (int i = lane; i < 16 * 4 * 32 * 8; i += 32) {
        int j = i & 7;
        int l = (i >> 3) & 31;
        int g = (i >> 8) & 3;
        int c = i >> 10;
        Bs[i] = Whp[((((long long)c * 128) + (g * 32 + nt)) * 32 + l) * 8 + j];
    }
    __syncthreads();   // LDS writes visible before ds reads

    float cval[8];
#pragma unroll
    for (int j = 0; j < 8; ++j) cval[j] = 0.0f;

    for (int t = 0; t < T_; ++t) {
        const _Float16* __restrict__ hin  = (t & 1) ? hB : hA;
        _Float16* __restrict__       hout = (t & 1) ? hA : hB;
        const _Float16* __restrict__ Hrow = hin + row * H_;

        v8f acc[4] = {};   // i, f, g, o
        for (int kc = 0; kc < H_; kc += 32) {
            uint4 a0 = *(const uint4*)(Hrow + kc + koff);
            uint4 a1 = *(const uint4*)(Hrow + kc + 8 + koff);
            v8u ar;
            ar[0] = a0.x; ar[1] = a0.y; ar[2] = a0.z; ar[3] = a0.w;
            ar[4] = a1.x; ar[5] = a1.y; ar[6] = a1.z; ar[7] = a1.w;
            v16h A = as_v16h(ar);
            int c = kc >> 5;
#pragma unroll
            for (int g = 0; g < 4; ++g) {
                const uint4* bp =
                    (const uint4*)(Bs + (((c * 4 + g) * 32 + lane) * 8));
                uint4 b0 = bp[0], b1 = bp[1];
                v8u br;
                br[0] = b0.x; br[1] = b0.y; br[2] = b0.z; br[3] = b0.w;
                br[4] = b1.x; br[5] = b1.y; br[6] = b1.z; br[7] = b1.w;
                acc[g] = __builtin_amdgcn_wmma_f32_16x16x32_f16(
                    false, A, false, as_v16h(br), (short)0, acc[g], false, false);
            }
        }

#pragma unroll
        for (int j = 0; j < 8; ++j) {
            int b = m0 + j + hi * 8;
            const _Float16* __restrict__ xr = xz + ((long long)b * T_ + t) * G4;
            float zi = acc[0][j] + (float)xr[0 * H_ + hcol];
            float zf = acc[1][j] + (float)xr[1 * H_ + hcol];
            float zg = acc[2][j] + (float)xr[2 * H_ + hcol];
            float zo = acc[3][j] + (float)xr[3 * H_ + hcol];
            float ig = sigmoidf_(zi);
            float fg = sigmoidf_(zf);
            float gg = tanhf(zg);
            float og = sigmoidf_(zo);
            float cn = fg * cval[j] + ig * gg;
            cval[j] = cn;
            float hv = og * tanhf(cn);
            hout[b * H_ + hcol] = (_Float16)hv;
            seq_out[((long long)b * T_ + t) * H_ + hcol] = hv;
            if (t == T_ - 1) hlast[b * H_ + hcol] = hv;
        }

        // ---- device-wide step barrier (release arrive, acquire spin) ----
        __threadfence();
        if (lane == 0)
            __hip_atomic_fetch_add(&bar[t], 1, __ATOMIC_RELEASE,
                                   __HIP_MEMORY_SCOPE_AGENT);
        while (__hip_atomic_load(&bar[t], __ATOMIC_ACQUIRE,
                                 __HIP_MEMORY_SCOPE_AGENT) < NBLK) {
            __builtin_amdgcn_s_sleep(2);
        }
    }
}

// ---------------------------------------------------------------------------
// Workspace layout (bytes):
//   [0,        1 MB)  Wx packed f16 pairs   (262144 u32)
//   [1 MB,     3 MB)  Wh packed f16 pairs   (524288 u32)
//   [3 MB,    +64KB)  h ping buffer (f16, 64*512)
//   [.. ,    +128KB)  h pong buffer (f16)
//   [.. ,     +2KB )  per-step barrier counters (int[512])
//   [4 MB,  132 MB)   xz activations (f16, 32768 x 2048)
// ---------------------------------------------------------------------------
extern "C" void kernel_launch(void* const* d_in, const int* in_sizes, int n_in,
                              void* d_out, int out_size, void* d_ws,
                              size_t ws_size, hipStream_t stream) {
    const int*   seq  = (const int*)d_in[0];
    const float* emb  = (const float*)d_in[1];
    const float* Wx   = (const float*)d_in[2];
    const float* Wh   = (const float*)d_in[3];
    const float* bias = (const float*)d_in[4];

    char* ws = (char*)d_ws;
    unsigned int* Wxp = (unsigned int*)(ws);
    unsigned int* Whp = (unsigned int*)(ws + (1 << 20));
    _Float16*     hA  = (_Float16*)(ws + 3 * (1 << 20));
    _Float16*     hB  = (_Float16*)(ws + 3 * (1 << 20) + (64 << 10));
    int*          bar = (int*)(ws + 3 * (1 << 20) + (128 << 10));
    _Float16*     xz  = (_Float16*)(ws + (4 << 20));

    float* seq_out = (float*)d_out;
    float* hlast   = seq_out + (long long)B_ * T_ * H_;
    float* masko   = hlast + B_ * H_;

    pack_w_kernel<<<(262144 + 255) / 256, 256, 0, stream>>>(Wx, Wxp, E_);
    pack_w_kernel<<<(524288 + 255) / 256, 256, 0, stream>>>(Wh, Whp, H_);
    init_state_kernel<<<(B_ * H_ + 255) / 256, 256, 0, stream>>>(hA, bar);
    mask_kernel<<<(B_ * T_ + 255) / 256, 256, 0, stream>>>(seq, masko);

    xz_kernel<<<dim3(2048, 32), 32, 0, stream>>>(seq, emb, Wxp, bias, xz);

    lstm_persistent_kernel<<<NBLK, 32, 0, stream>>>(xz, Whp, hA, hB, seq_out,
                                                    hlast, bar);
}
